// HunyuanVideoSingleAttention_5437428596771
// MI455X (gfx1250) — matrix operand order repaired and checked by reading, not compiled, not run on hardware
//
#include <hip/hip_runtime.h>

// ---------------------------------------------------------------------------
// Problem constants (match reference)
// ---------------------------------------------------------------------------
constexpr int S_LEN   = 2304;        // sequence length
constexpr int D_MODEL = 3072;        // hidden dim = HEADS*HD
constexpr int HEADS   = 24;
constexpr int HD      = 128;         // head dim
constexpr int NQKV    = 3 * D_MODEL; // 9216
constexpr int BATCH   = 2;

typedef __bf16 bf16;
typedef bf16  v8bf  __attribute__((ext_vector_type(8)));
typedef bf16  v16bf __attribute__((ext_vector_type(16)));
typedef float v8f   __attribute__((ext_vector_type(8)));

// TDM descriptor vector types (per probe: uint32x4, int32x8, int32x4 ...)
typedef unsigned int tdm_v4u __attribute__((ext_vector_type(4)));
typedef int          tdm_v4i __attribute__((ext_vector_type(4)));
typedef int          tdm_v8i __attribute__((ext_vector_type(8)));

#if defined(__gfx1250__) && __has_builtin(__builtin_amdgcn_tensor_load_to_lds)
#define HAVE_TDM 1
#else
#define HAVE_TDM 0
#endif

__device__ __forceinline__ v8f v8f_zero() {
    v8f z;
#pragma unroll
    for (int i = 0; i < 8; ++i) z[i] = 0.0f;
    return z;
}

// Build a 16-element bf16 fragment from two 8-element (16B) chunks.
__device__ __forceinline__ v16bf mk_frag(const bf16* p0, const bf16* p1) {
    union U { v16bf v; v8bf h[2]; } u;
    u.h[0] = *reinterpret_cast<const v8bf*>(p0);
    u.h[1] = *reinterpret_cast<const v8bf*>(p1);
    return u.v;
}

__device__ __forceinline__ v8f wmma_bf16(v16bf a, v16bf b, v8f c) {
    // D(f32 16x16) = A(bf16 16x32) * B(bf16 32x16) + C
    return __builtin_amdgcn_wmma_f32_16x16x32_bf16(
        /*neg_a=*/false, a, /*neg_b=*/false, b,
        /*c_mod=*/(short)0, c, /*reuse_a=*/false, /*reuse_b=*/false);
}

#if HAVE_TDM
// Issue one 2D TDM tile load: global (byte address) -> LDS (byte offset).
// g1 carries data_size/dims/strides/padding per ISA 8.4 (D# group 1).
__device__ __forceinline__ void tdm_load(unsigned lds_off, unsigned long long ga,
                                         tdm_v8i g1) {
    tdm_v4u g0;
    g0[0] = 1u;                                       // count=1, no gather
    g0[1] = lds_off;                                  // lds_addr (bytes)
    g0[2] = (unsigned)(ga & 0xFFFFFFFFull);           // global_addr[31:0]
    g0[3] = (unsigned)((ga >> 32) & 0x01FFFFFFull)    // global_addr[56:32]
          | (2u << 30);                               // type=2 ("image")
    tdm_v4i z4 = {0, 0, 0, 0};
#if __has_include(<hip/amd_detail/amd_gfx1250_TDM.h>)
    tdm_v8i z8 = {0, 0, 0, 0, 0, 0, 0, 0};            // clang-23 6-arg form
    __builtin_amdgcn_tensor_load_to_lds(g0, g1, z4, z4, z8, 0);
#else
    __builtin_amdgcn_tensor_load_to_lds(g0, g1, z4, z4, 0);  // ROCm 7.2 5-arg
#endif
}
#endif

// ---------------------------------------------------------------------------
// Kernel 1: QKV projection  C[m][n] = hidden[m][:] . W[n][:] + bias[n]
//   M = B*S = 4608, N = 9216, K = 3072.  bf16 WMMA, fp32 accumulate.
//   Epilogue scatters into attention layouts:
//     q,k -> [b][h][s][d] (bf16 row-major), v -> [b][h][d][s] (bf16, transposed)
// ---------------------------------------------------------------------------
__global__ __launch_bounds__(128)
void qkv_gemm_kernel(const float* __restrict__ A,     // [4608][3072]
                     const float* __restrict__ W,     // [9216][3072]
                     const float* __restrict__ bias,  // [9216]
                     bf16* __restrict__ qb, bf16* __restrict__ kb,
                     bf16* __restrict__ vt)
{
    __shared__ bf16 As[128][40];   // A tile  128 rows x 32 k (+pad)
    __shared__ bf16 Bs[128][40];   // W rows  128 n    x 32 k (+pad)

    const int tid  = threadIdx.x;
    const int lane = tid & 31;
    const int wid  = tid >> 5;          // 4 waves
    const int wm   = wid >> 1;          // wave tile: 64x64
    const int wn   = wid & 1;
    const int l15  = lane & 15;
    const int half = (lane < 16) ? 0 : 1;

    const int n0 = blockIdx.x * 128;
    const int m0 = blockIdx.y * 128;

    // n0 is 128-aligned -> whole block tile lives in one matrix/head.
    const int mat   = n0 / D_MODEL;            // 0=q 1=k 2=v
    const int head  = (n0 % D_MODEL) / HD;
    const int bidx  = m0 / S_LEN;              // 2304 % 128 == 0
    const int sBase = m0 % S_LEN;

    v8f acc[4][4];
#pragma unroll
    for (int mi = 0; mi < 4; ++mi)
#pragma unroll
        for (int ni = 0; ni < 4; ++ni) acc[mi][ni] = v8f_zero();

    const int lrow = tid >> 3;   // 0..15
    const int lcg  = tid & 7;    // 0..7  (4 floats each)

    for (int k0 = 0; k0 < D_MODEL; k0 += 32) {
        // ---- stage fp32 tiles into LDS as bf16 ----
#pragma unroll
        for (int p = 0; p < 8; ++p) {
            const int r = lrow + 16 * p;
            const float4 av = *reinterpret_cast<const float4*>(
                &A[(size_t)(m0 + r) * D_MODEL + k0 + 4 * lcg]);
            As[r][4 * lcg + 0] = (bf16)av.x;
            As[r][4 * lcg + 1] = (bf16)av.y;
            As[r][4 * lcg + 2] = (bf16)av.z;
            As[r][4 * lcg + 3] = (bf16)av.w;
            const float4 bv = *reinterpret_cast<const float4*>(
                &W[(size_t)(n0 + r) * D_MODEL + k0 + 4 * lcg]);
            Bs[r][4 * lcg + 0] = (bf16)bv.x;
            Bs[r][4 * lcg + 1] = (bf16)bv.y;
            Bs[r][4 * lcg + 2] = (bf16)bv.z;
            Bs[r][4 * lcg + 3] = (bf16)bv.w;
        }
        __syncthreads();

        if (k0 + 32 < D_MODEL) {   // hint next tiles -> global_prefetch_b8
            __builtin_prefetch(&A[(size_t)(m0 + lrow) * D_MODEL + k0 + 32], 0, 1);
            __builtin_prefetch(&W[(size_t)(n0 + lrow) * D_MODEL + k0 + 32], 0, 1);
        }

        // ---- fragments + 16 WMMAs ----
        v16bf af[4], bfv[4];
#pragma unroll
        for (int mi = 0; mi < 4; ++mi) {
            const int r  = 64 * wm + 16 * mi + l15;
            const int ko = half * 8;      // A layout: K 0..7/16..23 vs 8..15/24..31
            af[mi] = mk_frag(&As[r][ko], &As[r][ko + 16]);
        }
#pragma unroll
        for (int ni = 0; ni < 4; ++ni) {
            const int c  = 64 * wn + 16 * ni + l15;
            const int ko = half * 8;      // B layout: K 0..7/16..23 vs 8..15/24..31
            bfv[ni] = mk_frag(&Bs[c][ko], &Bs[c][ko + 16]);
        }
#pragma unroll
        for (int mi = 0; mi < 4; ++mi)
#pragma unroll
            for (int ni = 0; ni < 4; ++ni)
                acc[mi][ni] = wmma_bf16(af[mi], bfv[ni], acc[mi][ni]);
        __syncthreads();
    }

    // ---- epilogue: bias add, bf16 convert, scatter into q/k/v layouts ----
    const int rofs = half * 8;
#pragma unroll
    for (int mi = 0; mi < 4; ++mi) {
#pragma unroll
        for (int ni = 0; ni < 4; ++ni) {
            const int dcol = 64 * wn + 16 * ni + l15;     // head-dim index 0..127
            const float bv = bias[n0 + dcol];
#pragma unroll
            for (int j = 0; j < 8; ++j) {
                const int s   = sBase + 64 * wm + 16 * mi + j + rofs;
                const float v = acc[mi][ni][j] + bv;
                const size_t bh = (size_t)bidx * HEADS + head;
                if (mat == 0)
                    qb[(bh * S_LEN + s) * HD + dcol] = (bf16)v;
                else if (mat == 1)
                    kb[(bh * S_LEN + s) * HD + dcol] = (bf16)v;
                else
                    vt[(bh * HD + dcol) * S_LEN + s] = (bf16)v;
            }
        }
    }
}

// ---------------------------------------------------------------------------
// Kernel 2: per-(b,h,s) RMSNorm over 128 dims + interleaved RoPE (s >= T),
//           in-place on qb/kb. Q additionally pre-scaled by (1/sqrt(128))*log2e
//           so attention uses exp2 directly.
// ---------------------------------------------------------------------------
__global__ __launch_bounds__(64)
void norm_rope_kernel(bf16* __restrict__ qb, bf16* __restrict__ kb,
                      const float* __restrict__ wqn, const float* __restrict__ wkn,
                      const float* __restrict__ cosT, const float* __restrict__ sinT,
                      const int* __restrict__ tlen)
{
    const int bid = blockIdx.x;
    const int s = bid % S_LEN;
    const int h = (bid / S_LEN) % HEADS;
    const int b = bid / (S_LEN * HEADS);
    const size_t base = (((size_t)b * HEADS + h) * S_LEN + s) * HD;

    const int t  = threadIdx.x;      // 0..63, handles dims 2t, 2t+1
    const int d0 = 2 * t, d1 = 2 * t + 1;

    float q0 = (float)qb[base + d0], q1 = (float)qb[base + d1];
    float k0 = (float)kb[base + d0], k1 = (float)kb[base + d1];

    float sq = q0 * q0 + q1 * q1;
    float sk = k0 * k0 + k1 * k1;
    sq += __shfl_xor(sq, 16); sk += __shfl_xor(sk, 16);
    sq += __shfl_xor(sq, 8);  sk += __shfl_xor(sk, 8);
    sq += __shfl_xor(sq, 4);  sk += __shfl_xor(sk, 4);
    sq += __shfl_xor(sq, 2);  sk += __shfl_xor(sk, 2);
    sq += __shfl_xor(sq, 1);  sk += __shfl_xor(sk, 1);

    __shared__ float red[4];
    const int wv = t >> 5;
    if ((t & 31) == 0) { red[wv] = sq; red[2 + wv] = sk; }
    __syncthreads();
    const float rq = rsqrtf((red[0] + red[1]) * (1.0f / HD) + 1e-6f);
    const float rk = rsqrtf((red[2] + red[3]) * (1.0f / HD) + 1e-6f);

    q0 *= rq * wqn[d0]; q1 *= rq * wqn[d1];
    k0 *= rk * wkn[d0]; k1 *= rk * wkn[d1];

    const int T = tlen[0];
    if (s >= T) {
        const int pos = s - T;
        const float c  = cosT[pos * (HD / 2) + t];
        const float sn = sinT[pos * (HD / 2) + t];
        float o0 = q0 * c - q1 * sn, o1 = q1 * c + q0 * sn;
        q0 = o0; q1 = o1;
        o0 = k0 * c - k1 * sn; o1 = k1 * c + k0 * sn;
        k0 = o0; k1 = o1;
    }

    const float pre = 0.08838834764831845f * 1.4426950408889634f; // scale*log2e
    qb[base + d0] = (bf16)(q0 * pre);
    qb[base + d1] = (bf16)(q1 * pre);
    kb[base + d0] = (bf16)k0;
    kb[base + d1] = (bf16)k1;
}

// ---------------------------------------------------------------------------
// Kernel 3: flash attention. Block = 4 waves, each wave owns 16 q rows
//           (q tile 64). Key tiles of 32, staged into LDS by the Tensor Data
//           Mover (double buffered, TENSORcnt), shared by all 4 waves.
//           TDM pad_enable inserts 16B per row so LDS fragment reads are
//           bank-conflict-free: K pitch 272B (136 elems), V pitch 80B (40).
// ---------------------------------------------------------------------------
__global__ __launch_bounds__(128)
void attn_kernel(const bf16* __restrict__ qb, const bf16* __restrict__ kb,
                 const bf16* __restrict__ vt, float* __restrict__ out)
{
    __shared__ bf16 Ks[2][32][136];   // [buf][key][dim]  (TDM-padded rows)
    __shared__ bf16 Vs[2][128][40];   // [buf][dim][key]  (TDM-padded rows)
    __shared__ bf16 Ps[4][16][40];    // per-wave 16x32 P tile (+pad)

    const int w    = threadIdx.x >> 5;
    const int lane = threadIdx.x & 31;
    const int l15  = lane & 15;
    const int half = (lane < 16) ? 0 : 1;

    const int h = blockIdx.y, b = blockIdx.z;
    const int q0r = blockIdx.x * 64 + w * 16;
    const size_t bh = (size_t)b * HEADS + h;
    const bf16* qrow = qb + bh * S_LEN * HD;
    const bf16* krow = kb + bh * S_LEN * HD;
    const bf16* vrow = vt + bh * HD * S_LEN;

#if HAVE_TDM
    // D# group1 for K tile: data_size=2B; tensor_dim0=HD, tensor_dim1=S;
    // tile 128x32; dim0 stride=HD. pad: every 64 dwords (+4 dwords) -> 272B pitch.
    const tdm_v8i g1K = { 0x10000 | (1 << 20) | (5 << 22) | (3 << 25),
                          HD << 16, S_LEN << 16, HD << 16, 32, HD, 0, 0 };
    // D# group1 for V tile: tensor_dim0=S, tensor_dim1=HD; tile 32x128;
    // dim0 stride=S. pad: every 16 dwords (+4 dwords) -> 80B pitch.
    const tdm_v8i g1V = { 0x10000 | (1 << 20) | (3 << 22) | (3 << 25),
                          S_LEN << 16, HD << 16, 32 << 16, HD, S_LEN, 0, 0 };
    const unsigned long long kga = (unsigned long long)(uintptr_t)krow;
    const unsigned long long vga = (unsigned long long)(uintptr_t)vrow;
#endif

    auto stage = [&](int tile, int buf) {
#if HAVE_TDM
        if (w == 0) {
            tdm_load((unsigned)(uintptr_t)&Ks[buf][0][0],
                     kga + (unsigned long long)tile * 32 * HD * 2, g1K);
            tdm_load((unsigned)(uintptr_t)&Vs[buf][0][0],
                     vga + (unsigned long long)tile * 32 * 2, g1V);
        }
#else
        const int t = threadIdx.x;
#pragma unroll
        for (int c = t; c < 512; c += 128) {            // K: 32 rows x 16 chunks
            const int key = c >> 4, off = (c & 15) * 8;
            *(v8bf*)&Ks[buf][key][off] =
                *(const v8bf*)&krow[(size_t)(tile * 32 + key) * HD + off];
        }
#pragma unroll
        for (int c = t; c < 512; c += 128) {            // V: 128 rows x 4 chunks
            const int dim = c >> 2, off = (c & 3) * 8;
            *(v8bf*)&Vs[buf][dim][off] =
                *(const v8bf*)&vrow[(size_t)dim * S_LEN + tile * 32 + off];
        }
#endif
    };

    stage(0, 0);   // prologue: first tile in flight

    // ---- resident Q fragments (A-matrix, 16x128 over 4 k-chunks) ----
    v16bf qf[4];
    {
        const int r  = q0r + l15;
        const int ko = half * 8;
#pragma unroll
        for (int kk = 0; kk < 4; ++kk)
            qf[kk] = mk_frag(&qrow[(size_t)r * HD + 32 * kk + ko],
                             &qrow[(size_t)r * HD + 32 * kk + ko + 16]);
    }

    v8f o[8];
#pragma unroll
    for (int nc = 0; nc < 8; ++nc) o[nc] = v8f_zero();
    float mrow[8], lrow[8];
#pragma unroll
    for (int j = 0; j < 8; ++j) { mrow[j] = -1.0e30f; lrow[j] = 0.0f; }

    constexpr int NT = S_LEN / 32;   // 72 key tiles
    for (int it = 0; it < NT; ++it) {
        const int buf = it & 1;
        if (it + 1 < NT) stage(it + 1, buf ^ 1);   // overlap next DMA
#if HAVE_TDM
        if (w == 0) {
            if (it + 1 < NT) __builtin_amdgcn_s_wait_tensorcnt((short)2);
            else             __builtin_amdgcn_s_wait_tensorcnt((short)0);
        }
#endif
        __syncthreads();   // [A] tile `it` resident in LDS for all waves

        // ---- scores: two 16-key column tiles from LDS ----
        v8f sc[2];
#pragma unroll
        for (int nt = 0; nt < 2; ++nt) {
            sc[nt] = v8f_zero();
            const bf16* kp = &Ks[buf][16 * nt + l15][0];
            const int ko = half * 8;
#pragma unroll
            for (int kk = 0; kk < 4; ++kk) {
                v16bf bv = mk_frag(kp + 32 * kk + ko, kp + 32 * kk + ko + 16);
                sc[nt] = wmma_bf16(qf[kk], bv, sc[nt]);
            }
        }

        // ---- online softmax (per-row reductions across 16-lane halves) ----
        float pv[2][8];
#pragma unroll
        for (int j = 0; j < 8; ++j) {
            float tmax = fmaxf(sc[0][j], sc[1][j]);
            tmax = fmaxf(tmax, __shfl_xor(tmax, 1));
            tmax = fmaxf(tmax, __shfl_xor(tmax, 2));
            tmax = fmaxf(tmax, __shfl_xor(tmax, 4));
            tmax = fmaxf(tmax, __shfl_xor(tmax, 8));
            const float mn = fmaxf(mrow[j], tmax);
            const float r  = exp2f(mrow[j] - mn);
            const float p0 = exp2f(sc[0][j] - mn);
            const float p1 = exp2f(sc[1][j] - mn);
            float ps = p0 + p1;
            ps += __shfl_xor(ps, 1);
            ps += __shfl_xor(ps, 2);
            ps += __shfl_xor(ps, 4);
            ps += __shfl_xor(ps, 8);
            lrow[j] = lrow[j] * r + ps;
            mrow[j] = mn;
            pv[0][j] = p0; pv[1][j] = p1;
#pragma unroll
            for (int nc = 0; nc < 8; ++nc) o[nc][j] *= r;
        }

        // ---- P: C-layout -> A-layout via per-wave LDS transpose scratch ----
#pragma unroll
        for (int nt = 0; nt < 2; ++nt)
#pragma unroll
            for (int j = 0; j < 8; ++j)
                Ps[w][j + half * 8][16 * nt + l15] = (bf16)pv[nt][j];
        __syncthreads();   // [B]
        v16bf pf;
        {
            const int ko = half * 8;
            pf = mk_frag(&Ps[w][l15][ko], &Ps[w][l15][ko + 16]);
        }

        // ---- P.V: 8 dim-chunks of 16 against LDS-resident transposed V ----
#pragma unroll
        for (int nc = 0; nc < 8; ++nc) {
            const bf16* vp = &Vs[buf][16 * nc + l15][half * 8];
            v16bf vf = mk_frag(vp, vp + 16);
            o[nc] = wmma_bf16(pf, vf, o[nc]);
        }
        __syncthreads();   // [C] all waves done with buf before it is refilled
    }

    // ---- epilogue: divide by l, write out [b][s][h*128+d] fp32 ----
#pragma unroll
    for (int j = 0; j < 8; ++j) {
        const float inv = 1.0f / lrow[j];
        const int srow  = q0r + j + half * 8;
        const size_t ob = ((size_t)b * S_LEN + srow) * (HEADS * HD) + h * HD;
#pragma unroll
        for (int nc = 0; nc < 8; ++nc)
            out[ob + 16 * nc + l15] = o[nc][j] * inv;
    }
}

// ---------------------------------------------------------------------------
// Host launch
// ---------------------------------------------------------------------------
extern "C" void kernel_launch(void* const* d_in, const int* in_sizes, int n_in,
                              void* d_out, int out_size, void* d_ws, size_t ws_size,
                              hipStream_t stream) {
    const float* hidden = (const float*)d_in[0];
    const float* w_qkv  = (const float*)d_in[1];
    const float* b_qkv  = (const float*)d_in[2];
    const float* wq     = (const float*)d_in[3];
    const float* wk     = (const float*)d_in[4];
    const float* cosT   = (const float*)d_in[5];
    const float* sinT   = (const float*)d_in[6];
    const int*   tlen   = (const int*)d_in[7];
    float* out = (float*)d_out;

    char* ws = (char*)d_ws;
    const size_t bufBytes = (size_t)BATCH * HEADS * S_LEN * HD * sizeof(bf16); // ~27 MB
    bf16* qb = (bf16*)(ws);
    bf16* kb = (bf16*)(ws + bufBytes);
    bf16* vt = (bf16*)(ws + 2 * bufBytes);

    dim3 g1(NQKV / 128, (BATCH * S_LEN) / 128);            // 72 x 36
    qkv_gemm_kernel<<<g1, 128, 0, stream>>>(hidden, w_qkv, b_qkv, qb, kb, vt);

    norm_rope_kernel<<<BATCH * HEADS * S_LEN, 64, 0, stream>>>(
        qb, kb, wq, wk, cosT, sinT, tlen);

    dim3 g3(S_LEN / 64, HEADS, BATCH);                     // 36 x 24 x 2
    attn_kernel<<<g3, 128, 0, stream>>>(qb, kb, vt, out);
}
